// Discriminator_20675972563128
// MI455X (gfx1250) — matrix operand order repaired and checked by reading, not compiled, not run on hardware
//
#include <hip/hip_runtime.h>
#include <cstdint>

#define B_TOT 4096
#define T_LEN 256
#define D_IN  64
#define H_DIM 128
#define K3H   384   // 3*H

typedef __attribute__((ext_vector_type(16))) __bf16 v16bf;
typedef __attribute__((ext_vector_type(8)))  float  v8f;

// Fast transcendentals: keep the sequential GRU chain off the IEEE-div slow path.
__device__ __forceinline__ float fast_rcp(float x) {
  return __builtin_amdgcn_rcpf(x);           // v_rcp_f32
}
__device__ __forceinline__ float fast_sigmoid(float x) {
  return fast_rcp(1.0f + __expf(-x));        // v_mul + v_exp + v_add + v_rcp
}
__device__ __forceinline__ float fast_tanh(float x) {
#if __has_builtin(__builtin_amdgcn_tanhf)
  return __builtin_amdgcn_tanhf(x);          // native v_tanh_f32 on gfx1250
#else
  // tanh(x) = 1 - 2/(1 + e^{2x}); saturates correctly at +-inf
  return 1.0f - 2.0f * fast_rcp(1.0f + __expf(2.0f * x));
#endif
}

// A-fragment (16-bit, 16x32): lane L holds row M=L&15; K = kb..kb+7 and kb+16..kb+23
// where kb = chunkBase + (L<16 ? 0 : 8). Two 16B LDS reads, memory order == packed pairs.
__device__ __forceinline__ v16bf load_a16(const __bf16* p) {
  v16bf a;
  *(uint4*)&a         = *(const uint4*)(p);
  *(((uint4*)&a) + 1) = *(const uint4*)(p + 16);
  return a;
}

// B-fragment (16-bit, 32x16): lane L holds column N=L&15, K = kb..kb+7 and kb+16..kb+23,
// kb = chunkBase + (L<16 ? 0 : 8). Built once from global f32 weights -> packed bf16.
__device__ __forceinline__ v16bf make_b_frag(const float* __restrict__ W,
                                             int col, int kbase, int lane) {
  const int kb = kbase + ((lane & 16) ? 8 : 0);
  v16bf b;
  __bf16* e = (__bf16*)&b;
#pragma unroll
  for (int v = 0; v < 8; ++v) {
    e[2 * v]     = (__bf16)W[(size_t)(kb + 2 * v + (v >= 4 ? 8 : 0))     * K3H + col];
    e[2 * v + 1] = (__bf16)W[(size_t)(kb + 2 * v + 1 + (v >= 4 ? 8 : 0)) * K3H + col];
  }
  return b;
}

__global__ __launch_bounds__(256) void gru_fused_kernel(
    const float* __restrict__ x, const int* __restrict__ tlen,
    const float* __restrict__ Wk, const float* __restrict__ Rk,
    const float* __restrict__ bias, const float* __restrict__ dw,
    const float* __restrict__ db, float* __restrict__ out)
{
  __shared__ __bf16 x_lds[16 * D_IN];    // x_t tile, bf16
  __shared__ __bf16 h_lds[16 * H_DIM];   // shared h, bf16 (A operand for h@R)
  __shared__ float  s_part[16];
  __shared__ int    s_t[16];

  const int tid  = threadIdx.x;
  const int wave = tid >> 5;
  const int lane = tid & 31;
  const int n    = lane & 15;                 // tile column (B/C/D) and A-row
  const int rowh = (lane & 16) ? 8 : 0;       // C/D row offset for upper lane half
  const int kl   = (lane & 16) ? 8 : 0;       // A-fragment K sub-offset
  const int b0   = blockIdx.x * 16;

  // ---- per-row sequence lengths; zero h and dense partials ----
  if (tid < 16) {
    s_t[tid]    = tlen[b0 + tid];
    s_part[tid] = 0.0f;
  }
  for (int i = tid; i < 16 * H_DIM; i += 256) h_lds[i] = (__bf16)0.0f;
  __syncthreads();

  int tmax = 0;
#pragma unroll
  for (int i = 0; i < 16; ++i) tmax = max(tmax, s_t[i]);

  int t_row[8];
#pragma unroll
  for (int j = 0; j < 8; ++j) t_row[j] = s_t[rowh + j];

  // ---- register-resident bf16 weight B-fragments (one-time) ----
  // gate columns owned by this wave: z -> [0,128), r -> [128,256), h -> [256,384)
  const int cz = 0   + wave * 16 + n;
  const int cr = 128 + wave * 16 + n;
  const int ch = 256 + wave * 16 + n;

  v16bf wz[2], wr[2], wh[2];           // x-kernel, K=64 -> 2 chunks
#pragma unroll
  for (int c = 0; c < 2; ++c) {
    wz[c] = make_b_frag(Wk, cz, c * 32, lane);
    wr[c] = make_b_frag(Wk, cr, c * 32, lane);
    wh[c] = make_b_frag(Wk, ch, c * 32, lane);
  }
  v16bf rz[4], rr[4], rh[4];           // recurrent kernel, K=128 -> 4 chunks
#pragma unroll
  for (int c = 0; c < 4; ++c) {
    rz[c] = make_b_frag(Rk, cz, c * 32, lane);
    rr[c] = make_b_frag(Rk, cr, c * 32, lane);
    rh[c] = make_b_frag(Rk, ch, c * 32, lane);
  }

  // biases folded into accumulator init (reset_after=True: separate input/recurrent bias)
  const float bz  = bias[cz] + bias[K3H + cz];
  const float br  = bias[cr] + bias[K3H + cr];
  const float bxh = bias[ch];
  const float brh = bias[K3H + ch];

  float h_reg[8];                       // f32 h state for this wave's 16-column slab
#pragma unroll
  for (int j = 0; j < 8; ++j) h_reg[j] = 0.0f;

  // ---- software pipeline: x(t) prefetched into registers one step ahead ----
  const int xr = tid >> 4, xq = tid & 15;               // staging row / quad
  const float* xrow = x + ((size_t)(b0 + xr)) * T_LEN * D_IN + xq * 4;
  float4 xv;
  if (tmax > 0) {
    xv = *(const float4*)(xrow);                        // x(0)
  }

  for (int t = 0; t < tmax; ++t) {
    // publish prefetched x(t) tile -> LDS (bf16)
    {
      __bf16* dst = &x_lds[xr * D_IN + xq * 4];
      dst[0] = (__bf16)xv.x; dst[1] = (__bf16)xv.y;
      dst[2] = (__bf16)xv.z; dst[3] = (__bf16)xv.w;
    }
    __syncthreads();   // x_lds ready; h_lds writes from previous step visible

    // prefetch x(t+1): HBM latency hidden behind the 18 WMMAs below (uniform branch)
    if (t + 1 < tmax) {
      xv = *(const float4*)(xrow + (size_t)(t + 1) * D_IN);
    }

    v8f acc_z, acc_r, acc_xh, acc_rh;
#pragma unroll
    for (int j = 0; j < 8; ++j) {
      acc_z[j] = bz; acc_r[j] = br; acc_xh[j] = bxh; acc_rh[j] = brh;
    }

    // x_t @ W (K=64): 6 WMMAs
#pragma unroll
    for (int c = 0; c < 2; ++c) {
      v16bf a = load_a16(&x_lds[n * D_IN + c * 32 + kl]);
      acc_z  = __builtin_amdgcn_wmma_f32_16x16x32_bf16(false, a, false, wz[c], (short)0, acc_z,  false, false);
      acc_r  = __builtin_amdgcn_wmma_f32_16x16x32_bf16(false, a, false, wr[c], (short)0, acc_r,  false, false);
      acc_xh = __builtin_amdgcn_wmma_f32_16x16x32_bf16(false, a, false, wh[c], (short)0, acc_xh, false, false);
    }
    // h @ R (K=128): 12 WMMAs; rh accumulated separately (r gates only the recurrent term)
#pragma unroll
    for (int c = 0; c < 4; ++c) {
      v16bf a = load_a16(&h_lds[n * H_DIM + c * 32 + kl]);
      acc_z  = __builtin_amdgcn_wmma_f32_16x16x32_bf16(false, a, false, rz[c], (short)0, acc_z,  false, false);
      acc_r  = __builtin_amdgcn_wmma_f32_16x16x32_bf16(false, a, false, rr[c], (short)0, acc_r,  false, false);
      acc_rh = __builtin_amdgcn_wmma_f32_16x16x32_bf16(false, a, false, rh[c], (short)0, acc_rh, false, false);
    }

    // gates + masked state update (f32, hardware transcendentals)
#pragma unroll
    for (int j = 0; j < 8; ++j) {
      const float z  = fast_sigmoid(acc_z[j]);
      const float r  = fast_sigmoid(acc_r[j]);
      const float hh = fast_tanh(acc_xh[j] + r * acc_rh[j]);
      const float hn = z * h_reg[j] + (1.0f - z) * hh;
      h_reg[j] = (t < t_row[j]) ? hn : h_reg[j];
    }

    __syncthreads();   // all waves finished reading h_lds / x_lds this step

    // publish updated h slab (bf16) for next step's A operand
#pragma unroll
    for (int j = 0; j < 8; ++j) {
      h_lds[(rowh + j) * H_DIM + wave * 16 + n] = (__bf16)h_reg[j];
    }
    // next iteration's first barrier orders these writes before the reads
  }
  __syncthreads();

  // ---- dense head: logit = h @ dense_w + b ; y = sigmoid(logit) ----
  const float w_c = dw[wave * 16 + n];
#pragma unroll
  for (int j = 0; j < 8; ++j) {
    atomicAdd(&s_part[rowh + j], h_reg[j] * w_c);   // ds_add_f32
  }
  __syncthreads();
  if (tid < 16) {
    const float logit = s_part[tid] + db[0];
    out[b0 + tid]         = logit;                  // y_hat_logit
    out[B_TOT + b0 + tid] = fast_sigmoid(logit);    // y_hat
  }
}

extern "C" void kernel_launch(void* const* d_in, const int* in_sizes, int n_in,
                              void* d_out, int out_size, void* d_ws, size_t ws_size,
                              hipStream_t stream) {
  const float* x    = (const float*)d_in[0];
  const int*   t    = (const int*)  d_in[1];
  const float* Wk   = (const float*)d_in[2];
  const float* Rk   = (const float*)d_in[3];
  const float* bias = (const float*)d_in[4];
  const float* dw   = (const float*)d_in[5];
  const float* db   = (const float*)d_in[6];
  float* out = (float*)d_out;

  dim3 grid(B_TOT / 16);   // 256 blocks, 16 batch rows each
  dim3 block(256);         // 8 wave32s: one gate-column N-tile per wave
  gru_fused_kernel<<<grid, block, 0, stream>>>(x, t, Wk, Rk, bias, dw, db, out);
}
